// TransformerEncoder_43241730737000
// MI455X (gfx1250) — compile-verified, matrix operand order checked by reading, not run on hardware
//
#include <hip/hip_runtime.h>

// ---------------------------------------------------------------------------
// Transformer encoder (L=4, B=4, S=2048, D=1024) on gfx1250 via bf16 WMMA.
// 128x128 block GEMM, double-buffered LDS fed by GLOBAL_LOAD_ASYNC_TO_LDS
// (ASYNCcnt), compute core v_wmma_f32_16x16x32_bf16.
// ---------------------------------------------------------------------------

#define L_LAYERS 4
#define BATCH 4
#define SEQ 2048
#define DMODEL 1024
#define LDSTR 40   // padded LDS row stride (elements); 80B keeps 16B alignment

typedef __attribute__((ext_vector_type(16))) __bf16 v16bf;
typedef __attribute__((ext_vector_type(8)))  __bf16 v8bf;
typedef __attribute__((ext_vector_type(8)))  float  v8f;

__device__ __forceinline__ unsigned short f2bf(float f) {
    unsigned u = __float_as_uint(f);
    u += 0x7FFFu + ((u >> 16) & 1u);      // round-to-nearest-even
    return (unsigned short)(u >> 16);
}

// Wave-relative LDS byte offset of a generic pointer into a __shared__ array.
__device__ __forceinline__ unsigned lds_off(unsigned short* p) {
    return (unsigned)(unsigned long long)
        ((__attribute__((address_space(3))) unsigned short*)p);
}

// Async 16B global -> LDS copy (per lane), tracked by ASYNCcnt.
// ISA: GLOBAL_LOAD_ASYNC_TO_LDS_B128  vdst(LDS addr VGPR), vaddr(64b), off.
__device__ __forceinline__ void async_copy16(const unsigned short* g,
                                             unsigned short* l) {
    unsigned lofs = lds_off(l);
    asm volatile("global_load_async_to_lds_b128 %0, %1, off"
                 :: "v"(lofs), "v"(g) : "memory");
}

__device__ __forceinline__ void wait_async0() {
#if __has_builtin(__builtin_amdgcn_s_wait_asynccnt)
    __builtin_amdgcn_s_wait_asynccnt(0);
#else
    asm volatile("s_wait_asynccnt 0x0" ::: "memory");
#endif
}

// Assemble one 16-element bf16 WMMA operand for a lane from a row-major
// [row][k] tile. ISA 16-bit 16x32 layout: elements 0..7 <-> K = base+8h..+7,
// elements 8..15 <-> K = base+16+8h..+7. Caller pre-offsets p by row*ld + 8h.
__device__ __forceinline__ v16bf frag16(const unsigned short* p) {
    union { v16bf v; v8bf h[2]; } u;
    u.h[0] = *reinterpret_cast<const v8bf*>(p);
    u.h[1] = *reinterpret_cast<const v8bf*>(p + 16);
    return u.v;
}

// ---------------------------------------------------------------------------
// Block GEMM: C(128x128) = A(128xK, row-major, lda) * Bt(128xK, row-major,
// ldb)^T.  256 threads = 8 waves (4x2); each wave owns a 32x64 sub-tile.
// Async global->LDS double buffering, 32-deep K-steps.
// ---------------------------------------------------------------------------
__device__ __forceinline__ void block_gemm_128x128(
    const unsigned short* __restrict__ A, int lda,
    const unsigned short* __restrict__ Bt, int ldb,
    int m0, int n0, int K, v8f acc[2][4])
{
    __shared__ unsigned short As[2][128 * LDSTR];
    __shared__ unsigned short Bs[2][128 * LDSTR];

    const int tid  = threadIdx.x;
    const int lane = tid & 31;
    const int lr   = lane & 15;
    const int h    = lane >> 4;
    const int wave = tid >> 5;
    const int wy   = wave >> 1;   // 0..3  -> rows wy*32
    const int wx   = wave & 1;    // 0..1  -> cols wx*64

    // Cooperative staging: thread covers rows r0 and r0+64, 8 elems at kc.
    const int r0 = tid >> 2;             // 0..63
    const int kc = (tid & 3) * 8;        // 0,8,16,24
    const unsigned short* gA0 = A  + (size_t)(m0 + r0)      * lda + kc;
    const unsigned short* gA1 = A  + (size_t)(m0 + r0 + 64) * lda + kc;
    const unsigned short* gB0 = Bt + (size_t)(n0 + r0)      * ldb + kc;
    const unsigned short* gB1 = Bt + (size_t)(n0 + r0 + 64) * ldb + kc;
    const int lofs0 = r0 * LDSTR + kc;
    const int lofs1 = (r0 + 64) * LDSTR + kc;

    // Prologue: DMA chunk 0 into buffer 0.
    async_copy16(gA0, &As[0][lofs0]);
    async_copy16(gA1, &As[0][lofs1]);
    async_copy16(gB0, &Bs[0][lofs0]);
    async_copy16(gB1, &Bs[0][lofs1]);
    wait_async0();
    __syncthreads();

    const int nsteps = K >> 5;
    for (int s = 0; s < nsteps; ++s) {
        const int cur = s & 1;
        const int nxt = cur ^ 1;
        const bool more = (s + 1) < nsteps;
        if (more) {  // DMA next chunk while the matrix pipes run
            const int k = (s + 1) << 5;
            async_copy16(gA0 + k, &As[nxt][lofs0]);
            async_copy16(gA1 + k, &As[nxt][lofs1]);
            async_copy16(gB0 + k, &Bs[nxt][lofs0]);
            async_copy16(gB1 + k, &Bs[nxt][lofs1]);
        }
        // 8 WMMAs from LDS buffer `cur`.
        v16bf af[2], bf[4];
        #pragma unroll
        for (int mi = 0; mi < 2; ++mi)
            af[mi] = frag16(&As[cur][(wy * 32 + mi * 16 + lr) * LDSTR + 8 * h]);
        #pragma unroll
        for (int ni = 0; ni < 4; ++ni)
            bf[ni] = frag16(&Bs[cur][(wx * 64 + ni * 16 + lr) * LDSTR + 8 * h]);
        #pragma unroll
        for (int mi = 0; mi < 2; ++mi)
            #pragma unroll
            for (int ni = 0; ni < 4; ++ni)
                acc[mi][ni] = __builtin_amdgcn_wmma_f32_16x16x32_bf16(
                    false, af[mi], false, bf[ni], (short)0, acc[mi][ni],
                    false, false);
        if (more) {
            wait_async0();     // our DMA into `nxt` landed
            __syncthreads();   // everyone done reading `cur`, `nxt` visible
        }
    }
}

// ------------------------- data-prep kernels -------------------------------

__global__ __launch_bounds__(256) void k_cvt_x(const float* __restrict__ x,
                                               float* __restrict__ x32,
                                               unsigned short* __restrict__ xb,
                                               int n) {
    int i = blockIdx.x * 256 + threadIdx.x;
    if (i < n) { float f = x[i]; x32[i] = f; xb[i] = f2bf(f); }
}

__global__ __launch_bounds__(256) void k_tr_w(const float* __restrict__ W,
                                              unsigned short* __restrict__ Wt) {
    // Wt[l][f][d] = bf16(W[l][d][f])
    int i = blockIdx.x * 256 + threadIdx.x;
    int l = i / (DMODEL * DMODEL);
    int r = i - l * (DMODEL * DMODEL);
    int f = r / DMODEL;
    int d = r - f * DMODEL;
    Wt[i] = f2bf(W[(size_t)l * DMODEL * DMODEL + (size_t)d * DMODEL + f]);
}

__global__ __launch_bounds__(256) void k_tr_wo(const float* __restrict__ Wo,
                                               unsigned short* __restrict__ Wot) {
    // Wot[e][d] = bf16(Wo[d][e])
    int i = blockIdx.x * 256 + threadIdx.x;
    int e = i / DMODEL;
    int d = i - e * DMODEL;
    Wot[i] = f2bf(Wo[(size_t)d * DMODEL + e]);
}

__global__ __launch_bounds__(256) void k_tr_x(const unsigned short* __restrict__ xb,
                                              unsigned short* __restrict__ xbt) {
    // xbt[b][d][s] = xb[b][s][d]
    int i = blockIdx.x * 256 + threadIdx.x;
    int b = i / (DMODEL * SEQ);
    int r = i - b * (DMODEL * SEQ);
    int d = r / SEQ;
    int s = r - d * SEQ;
    xbt[i] = xb[(size_t)b * SEQ * DMODEL + (size_t)s * DMODEL + d];
}

// ------------------------- attention kernels -------------------------------

// scores[m][n] = sum_k xb[m][k] * xb[n][k]   (one batch; scale applied later)
__global__ __launch_bounds__(256) void k_scores(const unsigned short* __restrict__ xb,
                                                float* __restrict__ scores) {
    const int m0 = blockIdx.y * 128;
    const int n0 = blockIdx.x * 128;
    v8f acc[2][4] = {};
    block_gemm_128x128(xb, DMODEL, xb, DMODEL, m0, n0, DMODEL, acc);
    const int lane = threadIdx.x & 31;
    const int lr = lane & 15, h = lane >> 4;
    const int wave = threadIdx.x >> 5, wy = wave >> 1, wx = wave & 1;
    for (int mi = 0; mi < 2; ++mi)
        for (int ni = 0; ni < 4; ++ni)
            for (int r = 0; r < 8; ++r) {
                int row = m0 + wy * 32 + mi * 16 + r + 8 * h;
                int col = n0 + wx * 64 + ni * 16 + lr;
                scores[(size_t)row * SEQ + col] = acc[mi][ni][r];
            }
}

// Row-wise softmax of scale*scores -> bf16 probs.
__global__ __launch_bounds__(256) void k_softmax(const float* __restrict__ scores,
                                                 unsigned short* __restrict__ probs,
                                                 float scale) {
    const int row = blockIdx.x;
    const int tid = threadIdx.x;
    const float* srow = scores + (size_t)row * SEQ;
    unsigned short* prow = probs + (size_t)row * SEQ;
    float m = -3.4e38f;
    for (int j = tid; j < SEQ; j += 256) m = fmaxf(m, srow[j] * scale);
    for (int o = 16; o; o >>= 1) m = fmaxf(m, __shfl_xor(m, o, 32));
    __shared__ float sm[8];
    if ((tid & 31) == 0) sm[tid >> 5] = m;
    __syncthreads();
    float mall = sm[0];
    for (int w = 1; w < 8; ++w) mall = fmaxf(mall, sm[w]);
    float s = 0.f;
    for (int j = tid; j < SEQ; j += 256) s += __expf(srow[j] * scale - mall);
    for (int o = 16; o; o >>= 1) s += __shfl_xor(s, o, 32);
    __shared__ float ss[8];
    if ((tid & 31) == 0) ss[tid >> 5] = s;
    __syncthreads();
    float sall = 0.f;
    for (int w = 0; w < 8; ++w) sall += ss[w];
    const float inv = 1.0f / sall;
    for (int j = tid; j < SEQ; j += 256)
        prow[j] = f2bf(__expf(srow[j] * scale - mall) * inv);
}

// x_out = x_in + probs @ V, V supplied transposed as xbt[d][s]. One batch.
__global__ __launch_bounds__(256) void k_pv(const unsigned short* __restrict__ probs,
                                            const unsigned short* __restrict__ xbt,
                                            const float* __restrict__ x32in,
                                            float* __restrict__ x32out,
                                            unsigned short* __restrict__ xbout) {
    const int m0 = blockIdx.y * 128;
    const int n0 = blockIdx.x * 128;
    v8f acc[2][4] = {};
    block_gemm_128x128(probs, SEQ, xbt, SEQ, m0, n0, SEQ, acc);
    const int lane = threadIdx.x & 31;
    const int lr = lane & 15, h = lane >> 4;
    const int wave = threadIdx.x >> 5, wy = wave >> 1, wx = wave & 1;
    for (int mi = 0; mi < 2; ++mi)
        for (int ni = 0; ni < 4; ++ni)
            for (int r = 0; r < 8; ++r) {
                int row = m0 + wy * 32 + mi * 16 + r + 8 * h;
                int col = n0 + wx * 64 + ni * 16 + lr;
                size_t idx = (size_t)row * DMODEL + col;
                float v = acc[mi][ni][r] + x32in[idx];
                x32out[idx] = v;
                xbout[idx] = f2bf(v);
            }
}

// ------------------------------ FFN / output -------------------------------

// x_out = x_in + relu(xb @ W + b); W supplied transposed [f][d]. All batches.
__global__ __launch_bounds__(256) void k_ffn(const unsigned short* __restrict__ xb,
                                             const unsigned short* __restrict__ Wt,
                                             const float* __restrict__ bias,
                                             const float* __restrict__ x32in,
                                             float* __restrict__ x32out,
                                             unsigned short* __restrict__ xbout) {
    const int m0 = blockIdx.y * 128;
    const int n0 = blockIdx.x * 128;
    v8f acc[2][4] = {};
    block_gemm_128x128(xb, DMODEL, Wt, DMODEL, m0, n0, DMODEL, acc);
    const int lane = threadIdx.x & 31;
    const int lr = lane & 15, h = lane >> 4;
    const int wave = threadIdx.x >> 5, wy = wave >> 1, wx = wave & 1;
    for (int mi = 0; mi < 2; ++mi)
        for (int ni = 0; ni < 4; ++ni)
            for (int r = 0; r < 8; ++r) {
                int row = m0 + wy * 32 + mi * 16 + r + 8 * h;
                int col = n0 + wx * 64 + ni * 16 + lr;
                size_t idx = (size_t)row * DMODEL + col;
                float v = fmaxf(acc[mi][ni][r] + bias[col], 0.0f) + x32in[idx];
                x32out[idx] = v;
                xbout[idx] = f2bf(v);
            }
}

// out = xb @ Wo + bo; Wo supplied transposed [e][d]. All batches, f32 out.
__global__ __launch_bounds__(256) void k_outproj(const unsigned short* __restrict__ xb,
                                                 const unsigned short* __restrict__ Wot,
                                                 const float* __restrict__ bo,
                                                 float* __restrict__ out) {
    const int m0 = blockIdx.y * 128;
    const int n0 = blockIdx.x * 128;
    v8f acc[2][4] = {};
    block_gemm_128x128(xb, DMODEL, Wot, DMODEL, m0, n0, DMODEL, acc);
    const int lane = threadIdx.x & 31;
    const int lr = lane & 15, h = lane >> 4;
    const int wave = threadIdx.x >> 5, wy = wave >> 1, wx = wave & 1;
    for (int mi = 0; mi < 2; ++mi)
        for (int ni = 0; ni < 4; ++ni)
            for (int r = 0; r < 8; ++r) {
                int row = m0 + wy * 32 + mi * 16 + r + 8 * h;
                int col = n0 + wx * 64 + ni * 16 + lr;
                out[(size_t)row * DMODEL + col] = acc[mi][ni][r] + bo[col];
            }
}

// ------------------------------- launcher ----------------------------------

extern "C" void kernel_launch(void* const* d_in, const int* in_sizes, int n_in,
                              void* d_out, int out_size, void* d_ws, size_t ws_size,
                              hipStream_t stream) {
    (void)in_sizes; (void)n_in; (void)out_size; (void)ws_size;
    const float* x  = (const float*)d_in[0];
    const float* W  = (const float*)d_in[1];
    const float* bi = (const float*)d_in[2];
    const float* Wo = (const float*)d_in[3];
    const float* bo = (const float*)d_in[4];
    float* out = (float*)d_out;

    constexpr size_t NX   = (size_t)BATCH * SEQ * DMODEL;      // 8M elems
    constexpr size_t SZ32 = NX * 4;                             // 32 MB
    constexpr size_t SZ16 = NX * 2;                             // 16 MB
    unsigned char* ws = (unsigned char*)d_ws;
    float*          x32A   = (float*)(ws);
    float*          x32B   = (float*)(ws + SZ32);
    unsigned short* xbA    = (unsigned short*)(ws + 2 * SZ32);
    unsigned short* xbB    = (unsigned short*)(ws + 2 * SZ32 + SZ16);
    unsigned short* xbt    = (unsigned short*)(ws + 2 * SZ32 + 2 * SZ16);
    unsigned short* Wt     = (unsigned short*)(ws + 2 * SZ32 + 3 * SZ16);
    constexpr size_t SZWT  = (size_t)L_LAYERS * DMODEL * DMODEL * 2;   // 8 MB
    unsigned short* Wot    = (unsigned short*)(ws + 2 * SZ32 + 3 * SZ16 + SZWT);
    constexpr size_t SZWO  = (size_t)DMODEL * DMODEL * 2;               // 2 MB
    float*          scores = (float*)(ws + 2 * SZ32 + 3 * SZ16 + SZWT + SZWO);
    constexpr size_t SZSC  = (size_t)SEQ * SEQ * 4;                     // 16 MB
    unsigned short* probs  = (unsigned short*)(ws + 2 * SZ32 + 3 * SZ16 + SZWT + SZWO + SZSC);

    const float scale = 0.03125f;  // 1/sqrt(1024)

    k_cvt_x<<<(int)(NX / 256), 256, 0, stream>>>(x, x32A, xbA, (int)NX);
    k_tr_w<<<(L_LAYERS * DMODEL * DMODEL) / 256, 256, 0, stream>>>(W, Wt);
    k_tr_wo<<<(DMODEL * DMODEL) / 256, 256, 0, stream>>>(Wo, Wot);

    const dim3 gScores(SEQ / 128, SEQ / 128);            // 16 x 16
    const dim3 gPV(DMODEL / 128, SEQ / 128);             // 8 x 16
    const dim3 gFFN(DMODEL / 128, (BATCH * SEQ) / 128);  // 8 x 64

    for (int l = 0; l < L_LAYERS; ++l) {
        k_tr_x<<<(int)(NX / 256), 256, 0, stream>>>(xbA, xbt);
        for (int b = 0; b < BATCH; ++b) {
            const size_t xo = (size_t)b * SEQ * DMODEL;
            k_scores<<<gScores, 256, 0, stream>>>(xbA + xo, scores);
            k_softmax<<<SEQ, 256, 0, stream>>>(scores, probs, scale);
            k_pv<<<gPV, 256, 0, stream>>>(probs, xbt + (size_t)b * DMODEL * SEQ,
                                          x32A + xo, x32B + xo, xbB + xo);
        }
        k_ffn<<<gFFN, 256, 0, stream>>>(xbB, Wt + (size_t)l * DMODEL * DMODEL,
                                        bi + (size_t)l * DMODEL, x32B, x32A, xbA);
    }
    k_outproj<<<gFFN, 256, 0, stream>>>(xbA, Wot, bo, out);
}
